// PNPP_PointEnc_14972255994428
// MI455X (gfx1250) — compile-verified, matrix operand order checked by reading, not run on hardware
//
#include <hip/hip_runtime.h>
#include <hip/hip_bf16.h>

#define DEV __device__ __forceinline__

typedef __attribute__((ext_vector_type(16))) __bf16          v16bf;
typedef __attribute__((ext_vector_type(8)))  float           v8f;
typedef __attribute__((ext_vector_type(8)))  unsigned short  ushort8;
typedef __attribute__((ext_vector_type(16))) unsigned short  ushort16;

DEV unsigned short f2bf(float f) {
  unsigned u = __float_as_uint(f);
  u += 0x7FFFu + ((u >> 16) & 1u);           // round-to-nearest-even
  return (unsigned short)(u >> 16);
}
DEV float bf2f(unsigned short h) { return __uint_as_float(((unsigned)h) << 16); }

// ---------------------------------------------------------------------------
// Farthest point sampling: one block per batch, iterative argmax reduction.
// ---------------------------------------------------------------------------
__global__ void fps_kernel(const float* __restrict__ xyz, int N, int stride,
                           int npoint, int* __restrict__ idx) {
  int b = blockIdx.x;
  const float* base = xyz + (size_t)b * N * stride;
  int* out = idx + (size_t)b * npoint;
  __shared__ float sv[256];
  __shared__ int   si[256];
  __shared__ int   slast;
  float d[16];
#pragma unroll
  for (int k = 0; k < 16; k++) d[k] = 1e10f;
  if (threadIdx.x == 0) slast = 0;
  __syncthreads();
  for (int it = 0; it < npoint; ++it) {
    int last = slast;
    if (threadIdx.x == 0) out[it] = last;
    float px = base[last * stride + 0];
    float py = base[last * stride + 1];
    float pz = base[last * stride + 2];
    float best = -1.0f; int bi = 0; int k = 0;
    for (int i = threadIdx.x; i < N; i += 256, ++k) {
      float dx = base[i * stride + 0] - px;
      float dy = base[i * stride + 1] - py;
      float dz = base[i * stride + 2] - pz;
      float dd = dx * dx + dy * dy + dz * dz;
      float m = fminf(d[k], dd);
      d[k] = m;
      if (m > best) { best = m; bi = i; }
    }
    sv[threadIdx.x] = best; si[threadIdx.x] = bi;
    __syncthreads();
    for (int s = 128; s > 0; s >>= 1) {
      if (threadIdx.x < s) {
        if (sv[threadIdx.x + s] > sv[threadIdx.x]) {
          sv[threadIdx.x] = sv[threadIdx.x + s];
          si[threadIdx.x] = si[threadIdx.x + s];
        }
      }
      __syncthreads();
    }
    if (threadIdx.x == 0) slast = si[0];
    __syncthreads();
  }
}

// ---------------------------------------------------------------------------
// Gather xyz rows by index -> dense (B, npoint, 3)
// ---------------------------------------------------------------------------
__global__ void gather_xyz_kernel(const float* __restrict__ xyz, int stride, int N,
                                  const int* __restrict__ idx, int npoint,
                                  float* __restrict__ out, int total) {
  int t = blockIdx.x * 256 + threadIdx.x;
  if (t >= total) return;
  int c = t % 3; int r = t / 3;
  int b = r / npoint; int j = r % npoint;
  int s = idx[(size_t)b * npoint + j];
  out[t] = xyz[((size_t)b * N + s) * stride + c];
}

// ---------------------------------------------------------------------------
// Ball query: first nsample indices (ascending) within radius; pad with first.
// ---------------------------------------------------------------------------
__global__ void ball_query_kernel(const float* __restrict__ xyz, int stride, int N,
                                  const float* __restrict__ new_xyz, int npoint,
                                  float r2, int nsample, int* __restrict__ idx, int B) {
  int t = blockIdx.x * 256 + threadIdx.x;
  if (t >= B * npoint) return;
  int b = t / npoint;
  const float* src = xyz + (size_t)b * N * stride;
  float nx = new_xyz[t * 3 + 0], ny = new_xyz[t * 3 + 1], nz = new_xyz[t * 3 + 2];
  int* o = idx + (size_t)t * nsample;
  int cnt = 0, first = 0; bool found = false;
  for (int i = 0; i < N; i++) {
    float dx = src[i * stride + 0] - nx;
    float dy = src[i * stride + 1] - ny;
    float dz = src[i * stride + 2] - nz;
    float dd = dx * dx + dy * dy + dz * dz;
    if (dd < r2) {
      if (!found) { found = true; first = i; }
      o[cnt++] = i;
      if (cnt >= nsample) break;
    }
  }
  for (int k = cnt; k < nsample; k++) o[k] = first;
}

// ---------------------------------------------------------------------------
// Build grouped bf16 activation matrix: rows = B*npoint*nsample, cols = Kp.
// col < 3 : relative xyz;  3..3+C : gathered feats;  rest zero pad.
// ---------------------------------------------------------------------------
__global__ void group_kernel(const float* __restrict__ xyz, int xstride,
                             const float* __restrict__ feats, int fstride, int C,
                             const float* __restrict__ new_xyz,
                             const int* __restrict__ idx, int npoint, int nsample,
                             int Kp, unsigned short* __restrict__ A, int total, int N) {
  int t = blockIdx.x * 256 + threadIdx.x;
  if (t >= total) return;
  int c = t % Kp; int r = t / Kp;
  int s = r % nsample; int pj = r / nsample;
  int b = pj / npoint;
  int src = idx[(size_t)pj * nsample + s];
  float v = 0.0f;
  if (c < 3)            v = xyz[((size_t)b * N + src) * xstride + c] - new_xyz[(size_t)pj * 3 + c];
  else if (c < 3 + C)   v = feats[((size_t)b * N + src) * fstride + (c - 3)];
  A[t] = f2bf(v);
}

// ---------------------------------------------------------------------------
// Pack weights (cout x cin fp32) into WMMA B-fragment order, bf16:
// element index = ((kt*ntiles + nt)*32 + lane)*16 + e, zero-padded K.
// ---------------------------------------------------------------------------
__global__ void pack_weights_kernel(const float* __restrict__ W, int cout, int cin,
                                    int Kp, unsigned short* __restrict__ Wp, int total) {
  int t = blockIdx.x * 256 + threadIdx.x;
  if (t >= total) return;
  int e = t & 15; int lane = (t >> 4) & 31; int tile = t >> 9;
  int ntiles = cout >> 4;
  int kt = tile / ntiles, nt = tile % ntiles;
  int half = lane >> 4, n = lane & 15;
  int kin = (e < 8) ? (half * 8 + e) : (16 + half * 8 + (e - 8));
  int k = kt * 32 + kin;
  int col = nt * 16 + n;
  float v = (k < cin) ? W[(size_t)col * cin + k] : 0.0f;
  Wp[t] = f2bf(v);
}

// ---------------------------------------------------------------------------
// WMMA GEMM, N-register-blocked: one wave (32 threads) computes a 16x(16*NB)
// strip.  The A fragment is loaded ONCE per K-step and amortized over NB
// B-fragment loads + NB wmma ops (divides A HBM traffic by NB).
// C = relu(A(MxK,bf16,row-major lda=Kp) * packed-W + bias), bf16 out (+opt f32).
// ---------------------------------------------------------------------------
template <int NB>
__global__ void wmma_gemm_bias_relu_kernel(const unsigned short* __restrict__ A, int lda,
                                           const unsigned short* __restrict__ Wp,
                                           const float* __restrict__ bias, int Ncols,
                                           unsigned short* __restrict__ Cout, int ldc,
                                           float* __restrict__ Fout, int ldf) {
  int mt = blockIdx.x, ntg = blockIdx.y;       // group of NB consecutive N tiles
  int lane = threadIdx.x;
  int half = lane >> 4, lsub = lane & 15;
  int ktiles = lda >> 5;
  int ntiles = Ncols >> 4;
  const unsigned short* arow = A + ((size_t)(mt * 16 + lsub)) * lda + half * 8;
  v8f acc[NB];
#pragma unroll
  for (int nb = 0; nb < NB; ++nb) acc[nb] = v8f{};
  for (int kt = 0; kt < ktiles; ++kt) {
    const unsigned short* pa = arow + kt * 32;
    ushort8 alo = *(const ushort8*)pa;          // K = half*8 .. half*8+7
    ushort8 ahi = *(const ushort8*)(pa + 16);   // K = 16+half*8 ..
    ushort16 au;
#pragma unroll
    for (int i = 0; i < 8; i++) { au[i] = alo[i]; au[i + 8] = ahi[i]; }
    v16bf av = __builtin_bit_cast(v16bf, au);
    if (kt + 1 < ktiles) __builtin_prefetch(pa + 32, 0, 1);   // global_prefetch_b8
    const unsigned short* bbase =
        Wp + (((size_t)kt * ntiles + ntg * NB) * 32 + lane) * 16;
#pragma unroll
    for (int nb = 0; nb < NB; ++nb) {
      ushort16 bu = *(const ushort16*)(bbase + (size_t)nb * 512);
      v16bf bv = __builtin_bit_cast(v16bf, bu);
      acc[nb] = __builtin_amdgcn_wmma_f32_16x16x32_bf16(false, av, false, bv,
                                                        (short)0, acc[nb], false, false);
    }
  }
#pragma unroll
  for (int nb = 0; nb < NB; ++nb) {
    int col = (ntg * NB + nb) * 16 + lsub;
    float bb = bias[col];
#pragma unroll
    for (int vg = 0; vg < 8; ++vg) {
      int row = mt * 16 + half * 8 + vg;        // C/D layout: lanes16-31 -> M+8
      float v = fmaxf(acc[nb][vg] + bb, 0.0f);
      Cout[(size_t)row * ldc + col] = f2bf(v);
      if (Fout) Fout[(size_t)row * ldf + col] = v;
    }
  }
}

// ---------------------------------------------------------------------------
// Max-pool over nsample (bf16 in, f32 out feats)
// ---------------------------------------------------------------------------
__global__ void maxpool_kernel(const unsigned short* __restrict__ in, int ldc,
                               int nsample, int C, float* __restrict__ out, int total) {
  int t = blockIdx.x * 256 + threadIdx.x;
  if (t >= total) return;
  int c = t % C; int pj = t / C;
  float m = -1e30f;
  for (int s = 0; s < nsample; ++s)
    m = fmaxf(m, bf2f(in[((size_t)pj * nsample + s) * ldc + c]));
  out[t] = m;
}

// ---------------------------------------------------------------------------
// 3-NN search + inverse-distance weights.
// ---------------------------------------------------------------------------
__global__ void threenn_kernel(const float* __restrict__ xyz1, int s1, int N1,
                               const float* __restrict__ xyz2, int s2, int N2,
                               int* __restrict__ idx3, float* __restrict__ w3, int B) {
  int t = blockIdx.x * 256 + threadIdx.x;
  if (t >= B * N1) return;
  int b = t / N1;
  const float* p1 = xyz1 + (size_t)t * s1;
  float px = p1[0], py = p1[1], pz = p1[2];
  const float* src = xyz2 + (size_t)b * N2 * s2;
  float d0 = 1e30f, d1 = 1e30f, d2v = 1e30f; int i0 = 0, i1 = 0, i2 = 0;
  for (int i = 0; i < N2; i++) {
    float dx = src[i * s2 + 0] - px;
    float dy = src[i * s2 + 1] - py;
    float dz = src[i * s2 + 2] - pz;
    float dd = dx * dx + dy * dy + dz * dz;
    if (dd < d0)       { d2v = d1; i2 = i1; d1 = d0; i1 = i0; d0 = dd; i0 = i; }
    else if (dd < d1)  { d2v = d1; i2 = i1; d1 = dd; i1 = i; }
    else if (dd < d2v) { d2v = dd; i2 = i; }
  }
  float r0 = 1.0f / (d0 + 1e-8f), r1 = 1.0f / (d1 + 1e-8f), r2 = 1.0f / (d2v + 1e-8f);
  float s = 1.0f / (r0 + r1 + r2);
  idx3[(size_t)t * 3 + 0] = i0; idx3[(size_t)t * 3 + 1] = i1; idx3[(size_t)t * 3 + 2] = i2;
  w3[(size_t)t * 3 + 0] = r0 * s; w3[(size_t)t * 3 + 1] = r1 * s; w3[(size_t)t * 3 + 2] = r2 * s;
}

// ---------------------------------------------------------------------------
// FP concat: [feats1 | 3NN-interp(feats2)] -> bf16 activation rows, zero pad.
// ---------------------------------------------------------------------------
__global__ void concat_fp_kernel(const float* __restrict__ feats1, int f1s, int C1,
                                 const float* __restrict__ feats2, int C2, int N2,
                                 const int* __restrict__ idx3, const float* __restrict__ w3,
                                 int N1, int Kp, unsigned short* __restrict__ A, int total) {
  int t = blockIdx.x * 256 + threadIdx.x;
  if (t >= total) return;
  int c = t % Kp; int r = t / Kp;
  int b = r / N1;
  float v = 0.0f;
  if (c < C1) {
    v = feats1[(size_t)r * f1s + c];
  } else if (c < C1 + C2) {
    int cc = c - C1; float a = 0.0f;
#pragma unroll
    for (int k = 0; k < 3; k++) {
      int j = idx3[(size_t)r * 3 + k];
      a += w3[(size_t)r * 3 + k] * feats2[((size_t)b * N2 + j) * C2 + cc];
    }
    v = a;
  }
  A[t] = f2bf(v);
}

// ===========================================================================
// Host orchestration
// ===========================================================================
static inline int pad32(int x) { return (x + 31) & ~31; }

extern "C" void kernel_launch(void* const* d_in, const int* in_sizes, int n_in,
                              void* d_out, int out_size, void* d_ws, size_t ws_size,
                              hipStream_t stream) {
  (void)in_sizes; (void)n_in; (void)out_size; (void)ws_size;
  const int B = 16;
  const int Nlvl[5] = {4096, 1024, 256, 64, 16};
  const float radius[4] = {0.1f, 0.2f, 0.4f, 0.8f};
  const int NS = 32;

  struct LD { int cin, cout; };
  static const LD L[21] = {
    {6,32},{32,32},{32,64},        // SA1
    {67,64},{64,64},{64,128},      // SA2
    {131,128},{128,128},{128,256}, // SA3
    {259,256},{256,256},{256,512}, // SA4
    {768,256},{256,256},           // FP0
    {384,256},{256,256},           // FP1
    {320,256},{256,128},           // FP2
    {131,128},{128,128},{128,128}  // FP3
  };
  const int saL0[4] = {0, 3, 6, 9};
  const int fpL0[4] = {12, 14, 16, 18};
  const int fpNL[4] = {2, 2, 2, 3};

  const float* pc = (const float*)d_in[0];
  auto Wof = [&](int k) { return (const float*)d_in[1 + 2 * k]; };
  auto Bof = [&](int k) { return (const float*)d_in[2 + 2 * k]; };

  // ---- bump allocator over workspace ----
  char* ws = (char*)d_ws; size_t off = 0;
  auto alloc = [&](size_t bytes) -> void* {
    void* p = ws + off;
    off = (off + bytes + 255) & ~(size_t)255;
    return p;
  };

  unsigned short* Wp[21];
  for (int k = 0; k < 21; k++) {
    int Kp = pad32(L[k].cin);
    int total = Kp * L[k].cout;
    Wp[k] = (unsigned short*)alloc((size_t)total * 2);
    pack_weights_kernel<<<(total + 255) / 256, 256, 0, stream>>>(
        Wof(k), L[k].cout, L[k].cin, Kp, Wp[k], total);
  }

  unsigned short* bufA = (unsigned short*)alloc((size_t)16777216 * 2);
  unsigned short* bufB = (unsigned short*)alloc((size_t)33554432 * 2);

  float* xyzL[5]; float* featsL[5];
  xyzL[0] = nullptr; featsL[0] = nullptr;
  const int featsCap[5] = {0, 128, 256, 256, 512};
  for (int l = 1; l <= 4; l++) {
    xyzL[l]   = (float*)alloc((size_t)B * Nlvl[l] * 3 * 4);
    featsL[l] = (float*)alloc((size_t)B * Nlvl[l] * featsCap[l] * 4);
  }
  int*   idxF = (int*)alloc((size_t)B * 1024 * 4);
  int*   idxB = (int*)alloc((size_t)B * 1024 * NS * 4);
  int*   idx3 = (int*)alloc((size_t)B * 4096 * 3 * 4);
  float* w3   = (float*)alloc((size_t)B * 4096 * 3 * 4);

  // GEMM chain helper: input already in bufA; returns final bf16 buffer.
  auto run_mlp = [&](int first, int nlayers, int rows, float* foutLast, int ldfLast)
      -> unsigned short* {
    unsigned short* cur = bufA;
    for (int li = 0; li < nlayers; li++) {
      const LD& ly = L[first + li];
      unsigned short* nxt = (cur == bufA) ? bufB : bufA;
      float* fo = (li == nlayers - 1) ? foutLast : nullptr;
      int ntiles = ly.cout / 16;
      if (ntiles % 4 == 0) {
        dim3 grid(rows / 16, ntiles / 4);
        wmma_gemm_bias_relu_kernel<4><<<grid, 32, 0, stream>>>(
            cur, pad32(ly.cin), Wp[first + li], Bof(first + li), ly.cout,
            nxt, ly.cout, fo, ldfLast);
      } else {
        dim3 grid(rows / 16, ntiles / 2);
        wmma_gemm_bias_relu_kernel<2><<<grid, 32, 0, stream>>>(
            cur, pad32(ly.cin), Wp[first + li], Bof(first + li), ly.cout,
            nxt, ly.cout, fo, ldfLast);
      }
      cur = nxt;
    }
    return cur;
  };

  int chan[5] = {3, 64, 128, 256, 512};   // feature channels per level

  // -------------------- SA levels --------------------
  for (int l = 0; l < 4; l++) {
    int srcN = Nlvl[l];
    int np   = Nlvl[l + 1];
    const float* sxyz  = (l == 0) ? pc       : xyzL[l];
    int          sxs   = (l == 0) ? 6        : 3;
    const float* sfeat = (l == 0) ? (pc + 3) : featsL[l];
    int          sfs   = (l == 0) ? 6        : chan[l];
    int          C     = chan[l];

    fps_kernel<<<B, 256, 0, stream>>>(sxyz, srcN, sxs, np, idxF);
    {
      int total = B * np * 3;
      gather_xyz_kernel<<<(total + 255) / 256, 256, 0, stream>>>(
          sxyz, sxs, srcN, idxF, np, xyzL[l + 1], total);
    }
    {
      int total = B * np;
      ball_query_kernel<<<(total + 255) / 256, 256, 0, stream>>>(
          sxyz, sxs, srcN, xyzL[l + 1], np, radius[l] * radius[l], NS, idxB, B);
    }
    int Kp0 = pad32(3 + C);
    int rows = B * np * NS;
    {
      int total = rows * Kp0;
      group_kernel<<<(total + 255) / 256, 256, 0, stream>>>(
          sxyz, sxs, sfeat, sfs, C, xyzL[l + 1], idxB, np, NS, Kp0, bufA, total, srcN);
    }
    unsigned short* last = run_mlp(saL0[l], 3, rows, nullptr, 0);
    int coutL = L[saL0[l] + 2].cout;
    {
      int total = B * np * coutL;
      maxpool_kernel<<<(total + 255) / 256, 256, 0, stream>>>(
          last, coutL, NS, coutL, featsL[l + 1], total);
    }
    chan[l + 1] = coutL;
  }

  // -------------------- FP levels --------------------
  for (int j = 0; j < 4; j++) {
    int i = 4 - j;
    int N1 = Nlvl[i - 1], N2 = Nlvl[i];
    const float* xyz1 = (i - 1 == 0) ? pc : xyzL[i - 1];
    int s1 = (i - 1 == 0) ? 6 : 3;
    {
      int total = B * N1;
      threenn_kernel<<<(total + 255) / 256, 256, 0, stream>>>(
          xyz1, s1, N1, xyzL[i], 3, N2, idx3, w3, B);
    }
    int C1 = chan[i - 1], C2 = chan[i];
    int Kp = pad32(C1 + C2);
    const float* f1 = (i - 1 == 0) ? (pc + 3) : featsL[i - 1];
    int f1s = (i - 1 == 0) ? 6 : C1;
    int rows = B * N1;
    {
      int total = rows * Kp;
      concat_fp_kernel<<<(total + 255) / 256, 256, 0, stream>>>(
          f1, f1s, C1, featsL[i], C2, N2, idx3, w3, N1, Kp, bufA, total);
    }
    int nl = fpNL[j];
    int coutL = L[fpL0[j] + nl - 1].cout;
    float* fout = (i - 1 == 0) ? (float*)d_out : featsL[i - 1];
    run_mlp(fpL0[j], nl, rows, fout, coutL);
    chan[i - 1] = coutL;
  }
}
